// GIN_57939108823678
// MI455X (gfx1250) — matrix operand order repaired and checked by reading, not compiled, not run on hardware
//
#include <hip/hip_runtime.h>
#include <hip/hip_bf16.h>

typedef __attribute__((ext_vector_type(16))) __bf16 v16bf;
typedef __attribute__((ext_vector_type(8)))  __bf16 v8bf;
typedef __attribute__((ext_vector_type(4)))  __bf16 v4bf;
typedef __attribute__((ext_vector_type(8)))  float  v8f;

#define NNODES 200000
#define NEDGES 3200000
#define IN_F   128
#define HID    64
#define OUT_F  40
#define BN_EPS 1e-5f

// ---------------- f32 -> bf16 convert ----------------
__global__ void f32_to_bf16_kernel(const float* __restrict__ x,
                                   __bf16* __restrict__ y, int n) {
    int i = blockIdx.x * 256 + threadIdx.x;
    if (i < n) y[i] = (__bf16)x[i];
}

// ---------------- edge scatter: agg[dst] += h[src] ----------------
template<int F>
__global__ void scatter_kernel(const __bf16* __restrict__ h,
                               const int* __restrict__ src,
                               const int* __restrict__ dst,
                               float* __restrict__ agg) {
    const int GP = F / 4;
    int i = blockIdx.x * 256 + threadIdx.x;
    int e = i / GP;
    if (e >= NEDGES) return;
    int g = (i % GP) * 4;
    int s = src[e], d = dst[e];
    v4bf hv = *(const v4bf*)(h + (size_t)s * F + g);
    float* ap = agg + (size_t)d * F + g;
    unsafeAtomicAdd(ap + 0, (float)hv[0]);
    unsafeAtomicAdd(ap + 1, (float)hv[1]);
    unsafeAtomicAdd(ap + 2, (float)hv[2]);
    unsafeAtomicAdd(ap + 3, (float)hv[3]);
}

// ---------------- fused GIN layer: (1+eps)h + agg -> MLP (WMMA) -> BN -> ReLU ----------------
// Block: 128 threads (4 waves), 64 node rows per block, each wave owns a 16-row tile.
template<int F>
__global__ __launch_bounds__(128)
void gin_mlp_kernel(const __bf16* __restrict__ h_in, const float* __restrict__ agg,
                    const float* __restrict__ w1, const float* __restrict__ b1,
                    const float* __restrict__ w2, const float* __restrict__ b2,
                    const float* __restrict__ epsp,
                    const float* __restrict__ gamma, const float* __restrict__ beta,
                    const float* __restrict__ mean,  const float* __restrict__ var,
                    __bf16* __restrict__ h_out) {
    __shared__ __attribute__((aligned(32))) __bf16 ldsZ[64 * F];     // input tile (rows x F)
    __shared__ __attribute__((aligned(32))) __bf16 ldsW1T[64 * F];   // W1 transposed [n][k]
    __shared__ __attribute__((aligned(32))) __bf16 ldsH1[64 * 64];   // hidden tile
    __shared__ __attribute__((aligned(32))) __bf16 ldsW2T[64 * 64];  // W2 transposed [n][k]

    const int tid  = threadIdx.x;
    const int wave = tid >> 5;
    const int lane = tid & 31;
    const int mrow = lane & 15;
    const int half = lane >> 4;          // 0: lanes 0-15, 1: lanes 16-31
    const int rowBase = blockIdx.x * 64;
    const float eps1 = 1.0f + epsp[0];

    // Stage 1: Z = (1+eps)*h + agg  (bf16 into LDS), plus weights (transposed) into LDS
    for (int idx = tid; idx < 64 * F; idx += 128) {
        int r = idx / F, k = idx % F;
        int node = rowBase + r;
        float z = 0.0f;
        if (node < NNODES) {
            size_t off = (size_t)node * F + k;
            z = eps1 * (float)h_in[off] + agg[off];
        }
        ldsZ[idx] = (__bf16)z;
    }
    for (int idx = tid; idx < 64 * F; idx += 128) {
        int n = idx / F, k = idx % F;
        ldsW1T[idx] = (__bf16)w1[k * 64 + n];
    }
    for (int idx = tid; idx < 64 * 64; idx += 128) {
        int n = idx >> 6, k = idx & 63;
        ldsW2T[idx] = (__bf16)w2[k * 64 + n];
    }
    __syncthreads();

    // ---- GEMM1: [16 x F] @ [F x 64] with v_wmma_f32_16x16x32_bf16 ----
    v8f acc[4] = {};
    #pragma unroll
    for (int kt = 0; kt < F / 32; ++kt) {
        // A fragment: lane half selects contiguous K run [0..15] or [16..31]
        const __bf16* arow = &ldsZ[(wave * 16 + mrow) * F + kt * 32 + half * 16];
        v8bf alo = *(const v8bf*)(arow);
        v8bf ahi = *(const v8bf*)(arow + 8);
        v16bf a;
        #pragma unroll
        for (int i = 0; i < 8; ++i) { a[i] = alo[i]; a[i + 8] = ahi[i]; }
        #pragma unroll
        for (int nt = 0; nt < 4; ++nt) {
            // B fragment from transposed weights: contiguous 16 K-values per lane
            const __bf16* brow = &ldsW1T[(nt * 16 + mrow) * F + kt * 32 + half * 16];
            v16bf b = *(const v16bf*)(brow);
            acc[nt] = __builtin_amdgcn_wmma_f32_16x16x32_bf16(
                false, a, false, b, (short)0, acc[nt], false, false);
        }
    }

    // bias + ReLU -> H1 (bf16 in LDS). C layout: VGPR r -> M = r + 8*half, N = mrow
    #pragma unroll
    for (int nt = 0; nt < 4; ++nt) {
        int n = nt * 16 + mrow;
        float bb = b1[n];
        #pragma unroll
        for (int r = 0; r < 8; ++r) {
            int row = wave * 16 + half * 8 + r;
            float v = acc[nt][r] + bb;
            v = v > 0.0f ? v : 0.0f;
            ldsH1[row * 64 + n] = (__bf16)v;
        }
    }
    __syncthreads();

    // ---- GEMM2: [16 x 64] @ [64 x 64] ----
    v8f acc2[4] = {};
    #pragma unroll
    for (int kt = 0; kt < 2; ++kt) {
        const __bf16* arow = &ldsH1[(wave * 16 + mrow) * 64 + kt * 32 + half * 16];
        v8bf alo = *(const v8bf*)(arow);
        v8bf ahi = *(const v8bf*)(arow + 8);
        v16bf a;
        #pragma unroll
        for (int i = 0; i < 8; ++i) { a[i] = alo[i]; a[i + 8] = ahi[i]; }
        #pragma unroll
        for (int nt = 0; nt < 4; ++nt) {
            const __bf16* brow = &ldsW2T[(nt * 16 + mrow) * 64 + kt * 32 + half * 16];
            v16bf b = *(const v16bf*)(brow);
            acc2[nt] = __builtin_amdgcn_wmma_f32_16x16x32_bf16(
                false, a, false, b, (short)0, acc2[nt], false, false);
        }
    }

    // bias + BatchNorm + ReLU -> global bf16
    #pragma unroll
    for (int nt = 0; nt < 4; ++nt) {
        int n = nt * 16 + mrow;
        float bb = b2[n];
        float ga = gamma[n], be = beta[n], mu = mean[n];
        float inv = rsqrtf(var[n] + BN_EPS);
        #pragma unroll
        for (int r = 0; r < 8; ++r) {
            int node = rowBase + wave * 16 + half * 8 + r;
            if (node < NNODES) {
                float v = acc2[nt][r] + bb;
                v = (v - mu) * inv * ga + be;
                v = v > 0.0f ? v : 0.0f;
                h_out[(size_t)node * 64 + n] = (__bf16)v;
            }
        }
    }
}

// ---------------- output head + log_softmax ----------------
__global__ __launch_bounds__(256)
void out_kernel(const __bf16* __restrict__ h, const float* __restrict__ wout,
                const float* __restrict__ bout, float* __restrict__ out) {
    __shared__ float ldsH[64 * 64];
    __shared__ float ldsWT[OUT_F * 64];
    __shared__ float ldsL[64 * OUT_F];
    int tid = threadIdx.x;
    int nodeBase = blockIdx.x * 64;
    for (int idx = tid; idx < 64 * 64; idx += 256) {
        int r = idx >> 6, k = idx & 63;
        int node = nodeBase + r;
        ldsH[idx] = (node < NNODES) ? (float)h[(size_t)node * 64 + k] : 0.0f;
    }
    for (int idx = tid; idx < OUT_F * 64; idx += 256) {
        int o = idx >> 6, k = idx & 63;
        ldsWT[idx] = wout[k * OUT_F + o];
    }
    __syncthreads();
    {
        int node = tid >> 2;
        int q = tid & 3;
        for (int j = 0; j < 10; ++j) {
            int o = q * 10 + j;
            float s = bout[o];
            #pragma unroll 8
            for (int k = 0; k < 64; ++k) s += ldsH[node * 64 + k] * ldsWT[o * 64 + k];
            ldsL[node * OUT_F + o] = s;
        }
    }
    __syncthreads();
    if (tid < 64) {
        int node = nodeBase + tid;
        if (node < NNODES) {
            float mx = -1e30f;
            for (int o = 0; o < OUT_F; ++o) mx = fmaxf(mx, ldsL[tid * OUT_F + o]);
            float s = 0.0f;
            for (int o = 0; o < OUT_F; ++o) s += __expf(ldsL[tid * OUT_F + o] - mx);
            float lse = mx + __logf(s);
            for (int o = 0; o < OUT_F; ++o)
                out[(size_t)node * OUT_F + o] = ldsL[tid * OUT_F + o] - lse;
        }
    }
}

extern "C" void kernel_launch(void* const* d_in, const int* in_sizes, int n_in,
                              void* d_out, int out_size, void* d_ws, size_t ws_size,
                              hipStream_t stream) {
    const float* x        = (const float*)d_in[0];
    const int*   src      = (const int*)d_in[1];
    const int*   dst      = (const int*)d_in[2];
    const float* w1_first = (const float*)d_in[3];
    const float* w1_rest  = (const float*)d_in[4];
    const float* b1       = (const float*)d_in[5];
    const float* w2       = (const float*)d_in[6];
    const float* b2       = (const float*)d_in[7];
    const float* eps      = (const float*)d_in[8];
    const float* gamma    = (const float*)d_in[9];
    const float* beta     = (const float*)d_in[10];
    const float* mean     = (const float*)d_in[11];
    const float* var      = (const float*)d_in[12];
    const float* wout     = (const float*)d_in[13];
    const float* bout     = (const float*)d_in[14];
    float* out = (float*)d_out;

    char* ws = (char*)d_ws;
    __bf16* hA  = (__bf16*)ws;                                          // 200000*128 bf16
    __bf16* hB  = (__bf16*)(ws + (size_t)NNODES * IN_F * 2);            // 200000*64 bf16
    float*  agg = (float*)(ws + (size_t)NNODES * IN_F * 2
                              + (size_t)NNODES * HID * 2);              // 200000*128 f32

    // ---- layer 0 (F = 128) ----
    hipMemsetAsync(agg, 0, (size_t)NNODES * IN_F * sizeof(float), stream);
    f32_to_bf16_kernel<<<(NNODES * IN_F + 255) / 256, 256, 0, stream>>>(x, hA, NNODES * IN_F);
    scatter_kernel<IN_F><<<(NEDGES * (IN_F / 4) + 255) / 256, 256, 0, stream>>>(hA, src, dst, agg);
    gin_mlp_kernel<IN_F><<<(NNODES + 63) / 64, 128, 0, stream>>>(
        hA, agg, w1_first, b1, w2, b2, eps, gamma, beta, mean, var, hB);

    // ---- layers 1..4 (F = 64), h ping-ponged in place ----
    for (int i = 1; i < 5; ++i) {
        hipMemsetAsync(agg, 0, (size_t)NNODES * HID * sizeof(float), stream);
        scatter_kernel<HID><<<(NEDGES * (HID / 4) + 255) / 256, 256, 0, stream>>>(hB, src, dst, agg);
        gin_mlp_kernel<HID><<<(NNODES + 63) / 64, 128, 0, stream>>>(
            hB, agg, w1_rest + (size_t)(i - 1) * HID * HID, b1 + i * HID,
            w2 + (size_t)i * HID * HID, b2 + i * HID, eps + i,
            gamma + i * HID, beta + i * HID, mean + i * HID, var + i * HID, hB);
    }

    out_kernel<<<(NNODES + 63) / 64, 256, 0, stream>>>(hB, wout, bout, out);
}
